// ResNet_63548336111871
// MI455X (gfx1250) — matrix operand order repaired and checked by reading, not compiled, not run on hardware
//
#include <hip/hip_runtime.h>
#include <hip/hip_bf16.h>

// ---------------------------------------------------------------------------
// Persistent 16-WGP cluster kernel, sequential residual-MLP scan:
//   x_{t+1} = x_t + W2*relu(W1*relu(W0*[x_t;u_t]+b0)+b1)+b2
//
// Partitioning (2 cluster syncs per step instead of 3):
//   L0: row-split   -> WGP w computes h0[64w:64w+64], publishes f16  (sync A)
//   L1: row-split   -> WGP w computes h1[64w:64w+64], kept in LOCAL LDS
//   L2: column-split-> dx_partial = W2[:,64w:64w+64] @ h1_local (no exchange
//       needed between L1 and L2!), publish 256-f32 partial slot  (sync B)
//   finalize: every WGP redundantly reduces the 16 partial slots and keeps
//       the full f32 state x in its own LDS -> no state broadcast barrier.
// All weights stay LDS-resident (224 KB/WGP = whole 3.5 MB f16 model across
// the 16-WGP cluster). Deterministic (no float atomics).
// ---------------------------------------------------------------------------

typedef __attribute__((ext_vector_type(16))) _Float16 v16h;
typedef __attribute__((ext_vector_type(8)))  _Float16 v8h;
typedef __attribute__((ext_vector_type(8)))  float    v8f;
typedef __attribute__((ext_vector_type(4)))  float    v4f;

#define NWG 16
#define TPB 128

#if __has_builtin(__builtin_amdgcn_global_load_async_to_lds_b128)
#define HAVE_ASYNC_LDS 1
// signature (from clang diagnostic): (int4 AS1*, int4 AS3*, imm, imm)
typedef int v4i __attribute__((vector_size(16)));
typedef __attribute__((address_space(1))) v4i gas_v4i;
typedef __attribute__((address_space(3))) v4i las_v4i;
#endif

// A-tile (16x32 f16) from LDS per ISA 7.12.2 16-bit A layout.
__device__ __forceinline__ v16h load_a_tile(const _Float16* w, int rowstride,
                                            int rtile, int kt, int lane) {
  const int m  = lane & 15;
  const int kb = kt * 32 + ((lane >> 4) << 3);
  const _Float16* p = w + (size_t)(rtile * 16 + m) * rowstride + kb;
  v16h a;
  *( v8h*)&a      = *(const v8h*)(p);
  *(((v8h*)&a)+1) = *(const v8h*)(p + 16);
  return a;
}

// B (32x16 f16): activation segment replicated into all 16 columns.
__device__ __forceinline__ v16h load_b_vec(const _Float16* act, int kt, int lane) {
  const _Float16* p = act + kt * 32 + ((lane >> 4) << 3);
  v16h b;
  *( v8h*)&b      = *(const v8h*)(p);
  *(((v8h*)&b)+1) = *(const v8h*)(p + 16);
  return b;
}

__device__ __forceinline__ void cluster_sync(unsigned* bar) {
  __builtin_amdgcn_fence(__ATOMIC_RELEASE, "agent");
  __syncthreads();
  if (__builtin_amdgcn_cluster_id_x() != 0) {
    __builtin_amdgcn_s_cluster_barrier();        // s_barrier_signal/-wait -3
  } else {
    if (threadIdx.x == 0) {
      unsigned arrive = atomicAdd(bar, 1u);
      unsigned target = (arrive / NWG + 1u) * NWG;
      while (atomicAdd(bar, 0u) < target) __builtin_amdgcn_s_sleep(1);
    }
    __syncthreads();
  }
  __builtin_amdgcn_fence(__ATOMIC_ACQUIRE, "agent");
}

__global__ void convert_weights_kernel(const float* __restrict__ W0,
                                       const float* __restrict__ W1,
                                       const float* __restrict__ W2,
                                       _Float16* __restrict__ W0h,
                                       _Float16* __restrict__ W1h,
                                       _Float16* __restrict__ W2h,
                                       unsigned* __restrict__ bar) {
  const size_t i0 = (size_t)blockIdx.x * blockDim.x + threadIdx.x;
  const size_t st = (size_t)gridDim.x * blockDim.x;
  for (size_t i = i0; i < (size_t)1024 * 512;  i += st) W0h[i] = (_Float16)W0[i];
  for (size_t i = i0; i < (size_t)1024 * 1024; i += st) W1h[i] = (_Float16)W1[i];
  for (size_t i = i0; i < (size_t)256 * 1024;  i += st) W2h[i] = (_Float16)W2[i];
  if (i0 == 0) bar[0] = 0u;
}

__global__ __launch_bounds__(TPB, 1)
void resnet_scan_kernel(const float* __restrict__ x0,
                        const float* __restrict__ U,
                        const float* __restrict__ b0,
                        const float* __restrict__ b1,
                        const float* __restrict__ b2,
                        const _Float16* __restrict__ W0h,
                        const _Float16* __restrict__ W1h,
                        const _Float16* __restrict__ W2h,
                        _Float16* __restrict__ h0h,     // 1024 f16 exchange buf
                        float* __restrict__ partial,    // 16 x 256 f32 slots
                        float* __restrict__ out,        // (T+1) x 256 f32
                        unsigned* __restrict__ bar,
                        int T) {
  extern __shared__ __align__(16) char smem_raw[];
  _Float16* sW0  = (_Float16*)smem_raw;       // 64 x 512        (64 KB)
  _Float16* sW1  = sW0 + 64 * 512;            // 64 x 1024       (128 KB)
  _Float16* sW2c = sW1 + 64 * 1024;           // 256 x 64 (cols) (32 KB)
  _Float16* sZ   = sW2c + 256 * 64;           // 512  : z = [x;u] f16
  _Float16* sH0  = sZ + 512;                  // 1024 : full h0
  _Float16* sH1  = sH0 + 1024;                // 64   : LOCAL h1 slice
  float*    sXf  = (float*)(sH1 + 64);        // 256  : full f32 state

  const int w    = blockIdx.x;
  const int lane = threadIdx.x & 31;
  const int wid  = threadIdx.x >> 5;

  // ---- one-time LDS weight fill ----
  {
    const v8h* s0 = (const v8h*)(W0h + (size_t)w * 64 * 512);
    const v8h* s1 = (const v8h*)(W1h + (size_t)w * 64 * 1024);
    v8h* d0 = (v8h*)sW0; v8h* d1 = (v8h*)sW1; v8h* d2 = (v8h*)sW2c;
    for (int i = threadIdx.x; i < 64 * 512 / 8;  i += TPB) d0[i] = s0[i];
    for (int i = threadIdx.x; i < 64 * 1024 / 8; i += TPB) d1[i] = s1[i];
    // W2 column slice: sW2c[r][c] = W2[r][64w + c], rowstride 64
    for (int i = threadIdx.x; i < 256 * 64 / 8; i += TPB) {
      const int r = i >> 3, c8 = i & 7;
      d2[i] = *(const v8h*)(W2h + (size_t)r * 1024 + w * 64 + c8 * 8);
    }
  }
  // ---- init full state + emit output row 0 (rows owned by this WGP) ----
  for (int i = threadIdx.x; i < 256; i += TPB) {
    float v = x0[i];
    sXf[i] = v;
    sZ[i]  = (_Float16)v;
    if ((i >> 4) == w) out[i] = v;
  }
  cluster_sync(bar);

  for (int t = 0; t < T; ++t) {
    // stage u_t into z; prefetch next step's u row
    for (int i = threadIdx.x; i < 256; i += TPB)
      sZ[256 + i] = (_Float16)U[(size_t)t * 256 + i];
    if (t + 1 < T && threadIdx.x < 8)
      __builtin_prefetch(U + (size_t)(t + 1) * 256 + threadIdx.x * 32, 0, 3);
    __syncthreads();

    // ---- layer 0: rows [64w,64w+64), K=512, 2 interleaved accumulators ----
    {
      v8f c0 = {}, c1 = {};
      #pragma unroll
      for (int kt = 0; kt < 16; kt += 2) {
        v16h a0 = load_a_tile(sW0, 512, wid, kt,     lane);
        v16h b0v = load_b_vec(sZ, kt,     lane);
        v16h a1 = load_a_tile(sW0, 512, wid, kt + 1, lane);
        v16h b1v = load_b_vec(sZ, kt + 1, lane);
        c0 = __builtin_amdgcn_wmma_f32_16x16x32_f16(false, a0, false, b0v,
                                                    (short)0, c0, false, false);
        c1 = __builtin_amdgcn_wmma_f32_16x16x32_f16(false, a1, false, b1v,
                                                    (short)0, c1, false, false);
      }
      v8f c = c0 + c1;
      if ((lane & 15) == 0) {
        const int gr = w * 64 + wid * 16 + (lane >> 4) * 8;
        v8h o;
        #pragma unroll
        for (int r = 0; r < 8; ++r) {
          float v = c[r] + b0[gr + r];
          o[r] = (_Float16)(v > 0.f ? v : 0.f);
        }
        *(v8h*)(h0h + gr) = o;
      }
    }
    cluster_sync(bar);   // ---- sync A: h0 published ----

    // gather full h0 (2 KB) into LDS — async global->LDS DMA (ASYNCcnt path)
#ifdef HAVE_ASYNC_LDS
    {
      __builtin_amdgcn_global_load_async_to_lds_b128(
          (gas_v4i*)(h0h + threadIdx.x * 8),
          (las_v4i*)(sH0 + threadIdx.x * 8), 0, 0);
#if __has_builtin(__builtin_amdgcn_s_wait_asynccnt)
      __builtin_amdgcn_s_wait_asynccnt(0);
#else
      asm volatile("s_wait_asynccnt 0x0" ::: "memory");
#endif
    }
#else
    for (int i = threadIdx.x; i < 1024 / 8; i += TPB)
      ((v8h*)sH0)[i] = ((const v8h*)h0h)[i];
#endif
    __syncthreads();

    // ---- layer 1: rows [64w,64w+64), K=1024, 2 accumulators; h1 stays local
    {
      v8f c0 = {}, c1 = {};
      #pragma unroll
      for (int kt = 0; kt < 32; kt += 2) {
        v16h a0 = load_a_tile(sW1, 1024, wid, kt,     lane);
        v16h b0v = load_b_vec(sH0, kt,     lane);
        v16h a1 = load_a_tile(sW1, 1024, wid, kt + 1, lane);
        v16h b1v = load_b_vec(sH0, kt + 1, lane);
        c0 = __builtin_amdgcn_wmma_f32_16x16x32_f16(false, a0, false, b0v,
                                                    (short)0, c0, false, false);
        c1 = __builtin_amdgcn_wmma_f32_16x16x32_f16(false, a1, false, b1v,
                                                    (short)0, c1, false, false);
      }
      v8f c = c0 + c1;
      if ((lane & 15) == 0) {
        const int gr = w * 64 + wid * 16 + (lane >> 4) * 8;  // bias row
        const int lr = wid * 16 + (lane >> 4) * 8;           // local row
        v8h o;
        #pragma unroll
        for (int r = 0; r < 8; ++r) {
          float v = c[r] + b1[gr + r];
          o[r] = (_Float16)(v > 0.f ? v : 0.f);
        }
        *(v8h*)(sH1 + lr) = o;                               // ds_store_b128
      }
    }
    __syncthreads();   // local only — no cluster sync between L1 and L2!

    // ---- layer 2 partial: dxp[256] = W2[:,64w:64w+64] @ h1_local ----
    // 16 row tiles of 16, K=64 (2 k-tiles); 4 independent tiles per wave.
    {
      #pragma unroll
      for (int rt = 0; rt < 4; ++rt) {
        const int tile = wid * 4 + rt;
        v8f c = {};
        v16h a0 = load_a_tile(sW2c, 64, tile, 0, lane);
        v16h bb0 = load_b_vec(sH1, 0, lane);
        c = __builtin_amdgcn_wmma_f32_16x16x32_f16(false, a0, false, bb0,
                                                   (short)0, c, false, false);
        v16h a1 = load_a_tile(sW2c, 64, tile, 1, lane);
        v16h bb1 = load_b_vec(sH1, 1, lane);
        c = __builtin_amdgcn_wmma_f32_16x16x32_f16(false, a1, false, bb1,
                                                   (short)0, c, false, false);
        if ((lane & 15) == 0) {
          const int base = w * 256 + tile * 16 + (lane >> 4) * 8;
          v4f p0, p1;
          #pragma unroll
          for (int r = 0; r < 4; ++r) { p0[r] = c[r]; p1[r] = c[r + 4]; }
          *(v4f*)(partial + base)     = p0;
          *(v4f*)(partial + base + 4) = p1;
        }
      }
    }
    cluster_sync(bar);   // ---- sync B: partials published ----

    // ---- finalize (redundant on every WGP): x += b2 + sum of 16 partials
    for (int i = threadIdx.x; i < 256; i += TPB) {
      float acc = sXf[i] + b2[i];
      #pragma unroll
      for (int j = 0; j < NWG; ++j) acc += partial[j * 256 + i];
      sXf[i] = acc;
      sZ[i]  = (_Float16)acc;
      if ((i >> 4) == w) out[(size_t)(t + 1) * 256 + i] = acc;  // owner rows
    }
    // next iteration's stage-u __syncthreads() covers sZ/sXf visibility
  }
}

extern "C" void kernel_launch(void* const* d_in, const int* in_sizes, int n_in,
                              void* d_out, int out_size, void* d_ws, size_t ws_size,
                              hipStream_t stream) {
  const float* x0 = (const float*)d_in[0];
  const float* U  = (const float*)d_in[1];
  const float* W0 = (const float*)d_in[2];
  const float* b0 = (const float*)d_in[3];
  const float* W1 = (const float*)d_in[4];
  const float* b1 = (const float*)d_in[5];
  const float* W2 = (const float*)d_in[6];
  const float* b2 = (const float*)d_in[7];
  const int T = in_sizes[1] / 256;

  char* ws = (char*)d_ws;                       // ~3.7 MB used
  _Float16* W0h   = (_Float16*)(ws);            // 1 MB
  _Float16* W1h   = (_Float16*)(ws + 1048576);  // 2 MB
  _Float16* W2h   = (_Float16*)(ws + 3145728);  // 512 KB
  _Float16* h0h   = (_Float16*)(ws + 3670016);  // 2 KB
  float*    part  = (float*)   (ws + 3672064);  // 16 KB
  unsigned* bar   = (unsigned*)(ws + 3688448);  // 4 B

  hipLaunchKernelGGL(convert_weights_kernel, dim3(512), dim3(256), 0, stream,
                     W0, W1, W2, W0h, W1h, W2h, bar);

  const size_t lds_bytes =
      (size_t)(64 * 512 + 64 * 1024 + 256 * 64 + 512 + 1024 + 64) * 2 + 256 * 4;
  (void)hipFuncSetAttribute(reinterpret_cast<const void*>(&resnet_scan_kernel),
                            hipFuncAttributeMaxDynamicSharedMemorySize,
                            (int)lds_bytes);
  hipLaunchKernelGGL(resnet_scan_kernel, dim3(NWG), dim3(TPB), lds_bytes, stream,
                     x0, U, b0, b1, b2, W0h, W1h, W2h, h0h, part,
                     (float*)d_out, bar, T);
}
